// MultiHeadDistanceLayer_85633057947958
// MI455X (gfx1250) — matrix operand order repaired and checked by reading, not compiled, not run on hardware
//
#include <hip/hip_runtime.h>
#include <hip/hip_bf16.h>
#include <math.h>

typedef __attribute__((ext_vector_type(16))) _Float16 v16h;
typedef __attribute__((ext_vector_type(8)))  float    v8f;

#define NUM_HEAD 8
#define HEAD_DIM 32
#define BATCH    4
#define SEQ_L    2048
#define DIM_IN   256
#define DIM_QK   (NUM_HEAD * HEAD_DIM)   // 256
#define KBLOCK   128                     // keys staged per async round
#define NBLOCKS  (SEQ_L / KBLOCK)        // 16

// ---------------------------------------------------------------------------
// Kernel 1: fused Q/K projection GEMM.
// M = B*L = 8192, N = 256, K = 256. One wave computes one 16x16 tile of BOTH
// Q and K (shares the A fragment of x), looping K in steps of 32 via
// v_wmma_f32_16x16x32_f16. Epilogue adds per-head bias and stores f16.
// ---------------------------------------------------------------------------
__global__ __launch_bounds__(256) void mhd_proj_kernel(
    const float* __restrict__ x,
    const float* __restrict__ Wq, const float* __restrict__ bq,
    const float* __restrict__ Wk, const float* __restrict__ bk,
    _Float16* __restrict__ qbuf, _Float16* __restrict__ kbuf)
{
    const int lane = threadIdx.x & 31;
    const int wave = threadIdx.x >> 5;
    const int tile = blockIdx.x * 8 + wave;          // exact: 8192 tiles total
    const int ntiles_n = DIM_QK / 16;                // 16
    const int mbase = (tile / ntiles_n) * 16;
    const int nbase = (tile % ntiles_n) * 16;
    const int half  = lane >> 4;
    const int l15   = lane & 15;

    v8f accq = {};
    v8f acck = {};

    for (int kb = 0; kb < DIM_IN; kb += 32) {
        // A fragment (x row-major): lane holds row M=l15;
        // element i -> K = half*16 + i
        v16h a;
        const float* arow = x + (size_t)(mbase + l15) * DIM_IN + kb;
        #pragma unroll
        for (int i = 0; i < 16; ++i) {
            const int k = half * 16 + i;
            a[i] = (_Float16)arow[k];
        }
        // B fragments (W row-major [K,N]): lane holds col N=nbase+l15;
        // element e -> K = kb + half*16 + e
        v16h bqf, bkf;
        const float* wq = Wq + (size_t)(kb + half * 16) * DIM_QK + nbase + l15;
        const float* wk = Wk + (size_t)(kb + half * 16) * DIM_QK + nbase + l15;
        #pragma unroll
        for (int e = 0; e < 16; ++e) {
            bqf[e] = (_Float16)wq[(size_t)e * DIM_QK];
            bkf[e] = (_Float16)wk[(size_t)e * DIM_QK];
        }
        accq = __builtin_amdgcn_wmma_f32_16x16x32_f16(false, a, false, bqf,
                                                      (short)0, accq, false, false);
        acck = __builtin_amdgcn_wmma_f32_16x16x32_f16(false, a, false, bkf,
                                                      (short)0, acck, false, false);
    }

    const int col = nbase + l15;
    const int h   = col >> 5;
    const float bqv = bq[h];
    const float bkv = bk[h];
    #pragma unroll
    for (int r = 0; r < 8; ++r) {
        const int row = mbase + half * 8 + r;
        qbuf[(size_t)row * DIM_QK + col] = (_Float16)(accq[r] + bqv);
        kbuf[(size_t)row * DIM_QK + col] = (_Float16)(acck[r] + bkv);
    }
}

// ---------------------------------------------------------------------------
// Async staging helper: each thread pulls 32 contiguous bytes of K-head data
// into LDS via two GLOBAL_LOAD_ASYNC_TO_LDS_B128 (ASYNCcnt-tracked; the
// instruction offset applies to BOTH the LDS and global address).
// ---------------------------------------------------------------------------
__device__ __forceinline__ void async_stage_32b(unsigned lds_off,
                                                const _Float16* gaddr)
{
    asm volatile(
        "global_load_async_to_lds_b128 %0, %1, off\n\t"
        "global_load_async_to_lds_b128 %0, %1, off offset:16"
        :: "v"(lds_off), "v"((unsigned long long)(size_t)gaddr)
        : "memory");
}

__device__ __forceinline__ void wait_asynccnt0()
{
    asm volatile("s_wait_asynccnt 0x0" ::: "memory");
}

// ---------------------------------------------------------------------------
// Kernel 2: streaming attention-distance with cooperative async K staging.
// All 8 waves of a block share one (b,h) and 8 consecutive q-tiles, so they
// consume identical K rows: stage 128 keys (8 KB) per round into LDS with
// double buffering (async loads overlap the previous round's compute), then
// each wave runs 8x (WMMA 16x16x32 + fused Gaussian-prior online softmax).
// ---------------------------------------------------------------------------
__global__ __launch_bounds__(256) void mhd_attn_kernel(
    const _Float16* __restrict__ qbuf, const _Float16* __restrict__ kbuf,
    const float* __restrict__ prior_mean, const float* __restrict__ prior_std,
    float* __restrict__ out)
{
    __shared__ _Float16 shk[2][KBLOCK * HEAD_DIM];   // 2 x 8 KB

    const int tid  = threadIdx.x;
    const int lane = tid & 31;
    const int wave = tid >> 5;
    const int QT   = SEQ_L / 16;                     // 128 q-tiles per (b,h)
    const int t    = blockIdx.x * 8 + wave;          // exact: 4096 waves total
    const int b    = t / (NUM_HEAD * QT);
    const int rem  = t % (NUM_HEAD * QT);
    const int h    = rem / QT;                       // uniform across block
    const int qbase = (rem % QT) * 16;
    const int half  = lane >> 4;
    const int l15   = lane & 15;

    const float mean  = prior_mean[0];
    const float istd  = 1.0f / prior_std[0];
    const float coef  = 0.3989422804014327f * istd;  // 1/(std*sqrt(2*pi))
    const float scale = 0.17677669529663687f;        // HEAD_DIM^-0.5

    // Per-thread staging assignment: row = tid/2, 32-byte half-row = tid&1.
    const int srow  = tid >> 1;
    const int shalf = tid & 1;
    const _Float16* gk0 =
        kbuf + (size_t)(b * SEQ_L + srow) * DIM_QK + h * HEAD_DIM + shalf * 16;
    const unsigned lds_row_off =
        (unsigned)(size_t)(&shk[0][0]) + (unsigned)(srow * 64 + shalf * 32);

    // Q A-fragment (Dh=32 == one WMMA K-step), loaded once.
    v16h a;
    const _Float16* qrow =
        qbuf + (size_t)(b * SEQ_L + qbase + l15) * DIM_QK + h * HEAD_DIM;
    #pragma unroll
    for (int i = 0; i < 16; ++i)
        a[i] = qrow[half * 16 + i];

    float rmax[8], rl[8], rd[8];
    #pragma unroll
    for (int r = 0; r < 8; ++r) { rmax[r] = -3.0e38f; rl[r] = 0.f; rd[r] = 0.f; }

    // Prologue: stage block 0 into buffer 0.
    async_stage_32b(lds_row_off, gk0);

    for (int jb = 0; jb < NBLOCKS; ++jb) {
        wait_asynccnt0();        // this wave's loads for block jb are in LDS
        __syncthreads();         // => every wave's loads for block jb are in

        // Overlap: issue block jb+1 into the other buffer while computing.
        if (jb + 1 < NBLOCKS) {
            async_stage_32b(lds_row_off + (unsigned)(((jb + 1) & 1) ? 8192 : 0)
                                        - (unsigned)((jb & 1) ? 0 : 0),
                            gk0 + (size_t)(jb + 1) * KBLOCK * DIM_QK);
        }

        const _Float16* kblk = &shk[jb & 1][0];
        #pragma unroll
        for (int jl = 0; jl < 8; ++jl) {
            // B-fragment from LDS: lane l15 = key col, contiguous 32 B/lane.
            v16h bf;
            const _Float16* krow = kblk + (jl * 16 + l15) * HEAD_DIM + half * 16;
            #pragma unroll
            for (int e = 0; e < 16; ++e) bf[e] = krow[e];

            v8f c = {};
            c = __builtin_amdgcn_wmma_f32_16x16x32_f16(false, a, false, bf,
                                                       (short)0, c, false, false);

            const float jcol = (float)(jb * KBLOCK + jl * 16 + l15);
            #pragma unroll
            for (int r = 0; r < 8; ++r) {
                const float qpos  = (float)(qbase + half * 8 + r);
                const float dist  = jcol - qpos;
                const float z     = (dist - mean) * istd;
                const float prior = coef * __expf(-0.5f * z * z);
                const float logit = c[r] * scale * prior;
                const float mnew  = fmaxf(rmax[r], logit);
                const float corr  = __expf(rmax[r] - mnew);
                const float e     = __expf(logit - mnew);
                rl[r]   = rl[r] * corr + e;
                rd[r]   = rd[r] * corr + e * dist;
                rmax[r] = mnew;
            }
        }
        __syncthreads();         // block jb's buffer free before re-issue
    }

    // Merge across the 16 lanes holding each row (xor masks 1..8 stay in-half).
    #pragma unroll
    for (int off = 1; off < 16; off <<= 1) {
        #pragma unroll
        for (int r = 0; r < 8; ++r) {
            const float om = __shfl_xor(rmax[r], off, 32);
            const float ol = __shfl_xor(rl[r],   off, 32);
            const float od = __shfl_xor(rd[r],   off, 32);
            const float mnew = fmaxf(rmax[r], om);
            const float c1 = __expf(rmax[r] - mnew);
            const float c2 = __expf(om - mnew);
            rl[r]   = rl[r] * c1 + ol * c2;
            rd[r]   = rd[r] * c1 + od * c2;
            rmax[r] = mnew;
        }
    }

    if (l15 == 0) {
        #pragma unroll
        for (int r = 0; r < 8; ++r) {
            const int row = qbase + half * 8 + r;
            out[(size_t)(b * SEQ_L + row) * NUM_HEAD + h] = rd[r] / rl[r];
        }
    }
}

// ---------------------------------------------------------------------------
// Launch. Inputs (setup_inputs order):
//   0:x [4,2048,256] f32   1:Wq [256,256] f32   2:bq [8] f32
//   3:Wk [256,256] f32     4:bk [8] f32         5:prior_mean [1]  6:prior_std [1]
// Output: [4,2048,8] f32. Workspace: qbuf/kbuf f16 (4 MB each).
// ---------------------------------------------------------------------------
extern "C" void kernel_launch(void* const* d_in, const int* in_sizes, int n_in,
                              void* d_out, int out_size, void* d_ws, size_t ws_size,
                              hipStream_t stream) {
    (void)in_sizes; (void)n_in; (void)out_size; (void)ws_size;
    const float* x  = (const float*)d_in[0];
    const float* Wq = (const float*)d_in[1];
    const float* bq = (const float*)d_in[2];
    const float* Wk = (const float*)d_in[3];
    const float* bk = (const float*)d_in[4];
    const float* pm = (const float*)d_in[5];
    const float* ps = (const float*)d_in[6];
    float* out = (float*)d_out;

    const size_t qk_elems = (size_t)BATCH * SEQ_L * DIM_QK;
    _Float16* qbuf = (_Float16*)d_ws;
    _Float16* kbuf = qbuf + qk_elems;

    mhd_proj_kernel<<<1024, 256, 0, stream>>>(x, Wq, bq, Wk, bk, qbuf, kbuf);
    mhd_attn_kernel<<<512, 256, 0, stream>>>(qbuf, kbuf, pm, ps, out);
}
